// RelativeMultiHeadAttention_56023553409322
// MI455X (gfx1250) — matrix operand order repaired and checked by reading, not compiled, not run on hardware
//
#include <hip/hip_runtime.h>

#define B_   4
#define T_   512
#define M_   512
#define D_   512
#define H_   8
#define HD_  64
#define S_   1024
#define SCALE_ 0.125f   // 1/sqrt(64)

typedef __attribute__((ext_vector_type(16))) __bf16 v16bf;
typedef __attribute__((ext_vector_type(8)))  float  v8f;

union FragU {
  v16bf v;
  uint4 q[2];
  unsigned short u[16];
};

__device__ __forceinline__ unsigned short f2bf(float f) {
  union { float f; unsigned int u; } cv; cv.f = f;
  unsigned int u = cv.u;
  u += 0x7FFFu + ((u >> 16) & 1u);   // round-to-nearest-even
  return (unsigned short)(u >> 16);
}

__device__ __forceinline__ v8f zero8() {
  v8f z;
#pragma unroll
  for (int i = 0; i < 8; ++i) z[i] = 0.f;
  return z;
}

// ---------------------------------------------------------------------------
// Elementwise helpers
// ---------------------------------------------------------------------------
__global__ void cvt_f32_bf16(const float* __restrict__ src,
                             unsigned short* __restrict__ dst, int n) {
  int i = blockIdx.x * blockDim.x + threadIdx.x;
  if (i < n) dst[i] = f2bf(src[i]);
}

// kv[b,s,d] = (s < M) ? mem[b,s,d] : x[b,s-M,d]  (bf16)
__global__ void concat_kv_bf16(const float* __restrict__ mem,
                               const float* __restrict__ x,
                               unsigned short* __restrict__ kv) {
  int i = blockIdx.x * blockDim.x + threadIdx.x;
  if (i >= B_ * S_ * D_) return;
  int d = i % D_;
  int s = (i / D_) % S_;
  int b = i / (D_ * S_);
  float v = (s < M_) ? mem[((size_t)b * M_ + s) * D_ + d]
                     : x[((size_t)b * T_ + (s - M_)) * D_ + d];
  kv[i] = f2bf(v);
}

// WT[col][row] = W[row][col]  (D x D), f32 -> bf16
__global__ void transpose_cvt_bf16(const float* __restrict__ W,
                                   unsigned short* __restrict__ WT) {
  int i = blockIdx.x * blockDim.x + threadIdx.x;
  if (i >= D_ * D_) return;
  int col = i % D_;
  int row = i / D_;
  WT[(size_t)col * D_ + row] = f2bf(W[i]);
}

// ---------------------------------------------------------------------------
// Generic NT WMMA GEMM (bf16 in, f32 accumulate):
//   C[z, m, n] = sum_k A[z, m, k] * B[z, n, k]    (+ bias[n])
// A row base = A + z*aZ + m*ldA ; B row base = B + z*bZ + n*ldB  (K contiguous)
// C element at Cv[z*cZ + m*cR + n*cC], written as bf16 or f32.
// block = 128 threads (4 waves); each wave does one 16x16 tile.
// grid = (M/16, N/64, Z)
// ---------------------------------------------------------------------------
__launch_bounds__(128)
__global__ void gemm_nt_bf16(const unsigned short* __restrict__ A, long ldA, long aZ,
                             const unsigned short* __restrict__ Bm, long ldB, long bZ,
                             void* __restrict__ Cv, long cR, long cC, long cZ,
                             int K, int outF32, const float* __restrict__ bias) {
  const int wave  = threadIdx.x >> 5;
  const int lane  = threadIdx.x & 31;
  const int l16   = lane & 15;
  const int half  = lane >> 4;
  const int mBase = blockIdx.x * 16;
  const int nBase = (blockIdx.y * 4 + wave) * 16;
  const size_t z  = blockIdx.z;

  const unsigned short* Ap = A + z * (size_t)aZ + (size_t)(mBase + l16) * ldA;
  const unsigned short* Bp = Bm + z * (size_t)bZ + (size_t)(nBase + l16) * ldB;

  v8f acc = zero8();
  for (int k = 0; k < K; k += 32) {
    FragU fa, fb;
    // A fragment (16x32 bf16): lane row = l16; K chunk = half*16 (16 contig)
    fa.q[0] = *(const uint4*)(Ap + k + half * 16);
    fa.q[1] = *(const uint4*)(Ap + k + half * 16 + 8);
    // B fragment (32x16 bf16, NT): lane col = l16; K chunk = half*16 (16 contig)
    const uint4* pb = (const uint4*)(Bp + k + half * 16);
    fb.q[0] = pb[0];
    fb.q[1] = pb[1];
    acc = __builtin_amdgcn_wmma_f32_16x16x32_bf16(false, fa.v, false, fb.v,
                                                  (short)0, acc, false, false);
  }

  const int n = nBase + l16;
  const size_t cbase = z * (size_t)cZ + (size_t)n * cC;
#pragma unroll
  for (int r = 0; r < 8; ++r) {
    const int m = mBase + half * 8 + r;    // C layout: VGPR r -> M = r + half*8
    float v = acc[r];
    if (bias) v += bias[n];
    const size_t idx = cbase + (size_t)m * cR;
    if (outF32) ((float*)Cv)[idx] = v;
    else        ((unsigned short*)Cv)[idx] = f2bf(v);
  }
}

// ---------------------------------------------------------------------------
// Relative-position scores, restructured:
//   P[t, bh, s] = sum_d Qt[b,t,h,d] * rel[t,s,d]
// One workgroup per t (512 blocks, 8 waves). rel_pos is streamed exactly once
// (1.07 GB total -> HBM roofline ~46us), converted f32->bf16 in-register.
// ---------------------------------------------------------------------------
__launch_bounds__(256)
__global__ void pos_scores(const unsigned short* __restrict__ Qt,
                           const float* __restrict__ rel,
                           float* __restrict__ P) {
  const int t    = blockIdx.x;
  const int wave = threadIdx.x >> 5;
  const int lane = threadIdx.x & 31;
  const int l16  = lane & 15;
  const int half = lane >> 4;

  // A-row bases for the lane's rows in both m-tiles (bh = mt*16 + l16)
  size_t abase[2];
#pragma unroll
  for (int mt = 0; mt < 2; ++mt) {
    int bh = mt * 16 + l16;
    int b = bh >> 3, h = bh & 7;
    abase[mt] = ((size_t)(b * T_ + t) * H_ + h) * D_;
  }

  for (int nt = wave; nt < S_ / 16; nt += 8) {
    const int s0 = nt * 16;
    const float* brow = rel + ((size_t)t * S_ + s0 + l16) * D_;
    v8f acc0 = zero8(), acc1 = zero8();
    for (int k = 0; k < D_; k += 32) {
      FragU fb;
      const float4* pf = (const float4*)(brow + k + half * 16);
#pragma unroll
      for (int q4 = 0; q4 < 4; ++q4) {
        float4 f = pf[q4];
        fb.u[q4 * 4 + 0] = f2bf(f.x);
        fb.u[q4 * 4 + 1] = f2bf(f.y);
        fb.u[q4 * 4 + 2] = f2bf(f.z);
        fb.u[q4 * 4 + 3] = f2bf(f.w);
      }
      FragU fa0, fa1;
      fa0.q[0] = *(const uint4*)(Qt + abase[0] + k + half * 16);
      fa0.q[1] = *(const uint4*)(Qt + abase[0] + k + half * 16 + 8);
      fa1.q[0] = *(const uint4*)(Qt + abase[1] + k + half * 16);
      fa1.q[1] = *(const uint4*)(Qt + abase[1] + k + half * 16 + 8);
      acc0 = __builtin_amdgcn_wmma_f32_16x16x32_bf16(false, fa0.v, false, fb.v,
                                                     (short)0, acc0, false, false);
      acc1 = __builtin_amdgcn_wmma_f32_16x16x32_bf16(false, fa1.v, false, fb.v,
                                                     (short)0, acc1, false, false);
    }
#pragma unroll
    for (int r = 0; r < 8; ++r) {
      const int bh0 = half * 8 + r;
      const int bh1 = 16 + half * 8 + r;
      P[((size_t)t * (B_ * H_) + bh0) * S_ + s0 + l16] = acc0[r];
      P[((size_t)t * (B_ * H_) + bh1) * S_ + s0 + l16] = acc1[r];
    }
  }
}

// ---------------------------------------------------------------------------
// Attention core: one wave per (b, h, 16-row t-tile). Flash-style online
// softmax over s in chunks of 32. Content logits via WMMA (K = HD = 64),
// position term added from P, probs(bf16) @ V^T accumulated via WMMA.
// Vt layout: [B][D][S] so the B-fragment is contiguous in s.
// ---------------------------------------------------------------------------
__launch_bounds__(32)
__global__ void attn_core(const unsigned short* __restrict__ Qb,
                          const unsigned short* __restrict__ Kb,
                          const unsigned short* __restrict__ Vt,
                          const float* __restrict__ P,
                          unsigned short* __restrict__ AttnOut) {
  __shared__ float ls[16][32];
  __shared__ unsigned short pb[16][32];
  __shared__ float fac[16];
  __shared__ float rsum[16];

  const int t0   = blockIdx.x * 16;
  const int h    = blockIdx.y;
  const int b    = blockIdx.z;
  const int lane = threadIdx.x & 31;
  const int l16  = lane & 15;
  const int half = lane >> 4;
  const int bh   = b * H_ + h;

  v8f oacc[4];
#pragma unroll
  for (int dt = 0; dt < 4; ++dt) oacc[dt] = zero8();
  float runmax = -3.0e38f, runsum = 0.f;

  const unsigned short* qrowp =
      Qb + (size_t)(b * T_ + t0 + l16) * D_ + h * HD_;

  for (int s0 = 0; s0 < S_; s0 += 32) {
    // ---- logits chunk: content WMMA + pos from P, scaled ----
    for (int nt = 0; nt < 2; ++nt) {
      const unsigned short* krowp =
          Kb + (size_t)(b * S_ + s0 + nt * 16 + l16) * D_ + h * HD_;
      v8f acc = zero8();
      for (int k = 0; k < HD_; k += 32) {
        FragU fa, fb;
        fa.q[0] = *(const uint4*)(qrowp + k + half * 16);
        fa.q[1] = *(const uint4*)(qrowp + k + half * 16 + 8);
        const uint4* pk = (const uint4*)(krowp + k + half * 16);
        fb.q[0] = pk[0];
        fb.q[1] = pk[1];
        acc = __builtin_amdgcn_wmma_f32_16x16x32_bf16(false, fa.v, false, fb.v,
                                                      (short)0, acc, false, false);
      }
      const int scol = nt * 16 + l16;
#pragma unroll
      for (int r = 0; r < 8; ++r) {
        const int m = half * 8 + r;
        const float pos =
            P[((size_t)(t0 + m) * (B_ * H_) + bh) * S_ + s0 + scol];
        ls[m][scol] = (acc[r] + pos) * SCALE_;
      }
    }
    __syncthreads();

    // ---- online softmax update (one row per thread, lanes 0..15) ----
    if (lane < 16) {
      const int r = lane;
      float cm = -3.0e38f;
      for (int c = 0; c < 32; ++c) cm = fmaxf(cm, ls[r][c]);
      const float nm = fmaxf(runmax, cm);
      const float corr = __expf(runmax - nm);
      float s = 0.f;
      for (int c = 0; c < 32; ++c) {
        float e = __expf(ls[r][c] - nm);
        s += e;
        pb[r][c] = f2bf(e);
      }
      runsum = runsum * corr + s;
      runmax = nm;
      fac[r] = corr;
    }
    __syncthreads();

    // ---- rescale accumulators, then probs(16x32) @ Vt(32x64) ----
    FragU fa;
#pragma unroll
    for (int j = 0; j < 8; ++j) {
      fa.u[j]     = pb[l16][half * 16 + j];
      fa.u[8 + j] = pb[l16][half * 16 + 8 + j];
    }
#pragma unroll
    for (int dt = 0; dt < 4; ++dt) {
#pragma unroll
      for (int r = 0; r < 8; ++r) oacc[dt][r] *= fac[half * 8 + r];
      FragU fb;
      const uint4* pv = (const uint4*)(Vt +
          ((size_t)b * D_ + h * HD_ + dt * 16 + l16) * S_ + s0 + half * 16);
      fb.q[0] = pv[0];
      fb.q[1] = pv[1];
      oacc[dt] = __builtin_amdgcn_wmma_f32_16x16x32_bf16(
          false, fa.v, false, fb.v, (short)0, oacc[dt], false, false);
    }
    __syncthreads();
  }

  if (lane < 16) rsum[lane] = runsum;
  __syncthreads();
#pragma unroll
  for (int dt = 0; dt < 4; ++dt) {
#pragma unroll
    for (int r = 0; r < 8; ++r) {
      const int m = half * 8 + r;
      const float v = oacc[dt][r] / rsum[m];
      AttnOut[(size_t)(b * T_ + t0 + m) * D_ + h * HD_ + dt * 16 + l16] =
          f2bf(v);
    }
  }
}

// ---------------------------------------------------------------------------
// Host launcher
// ---------------------------------------------------------------------------
extern "C" void kernel_launch(void* const* d_in, const int* in_sizes, int n_in,
                              void* d_out, int out_size, void* d_ws, size_t ws_size,
                              hipStream_t stream) {
  (void)in_sizes; (void)n_in; (void)out_size; (void)ws_size;
  const float* x   = (const float*)d_in[0];
  const float* mem = (const float*)d_in[1];
  const float* rel = (const float*)d_in[2];
  const float* Wq  = (const float*)d_in[3];
  const float* Wk  = (const float*)d_in[4];
  const float* Wv  = (const float*)d_in[5];
  const float* Wr  = (const float*)d_in[6];
  const float* Wo  = (const float*)d_in[7];
  const float* bo  = (const float*)d_in[8];
  float* out = (float*)d_out;

  char* ws = (char*)d_ws;
  size_t off = 0;
  auto take = [&](size_t bytes) -> char* {
    char* p = ws + off;
    off = (off + bytes + 255) & ~(size_t)255;
    return p;
  };
  unsigned short* kv_bf  = (unsigned short*)take((size_t)B_ * S_ * D_ * 2);
  unsigned short* q_bf   = (unsigned short*)take((size_t)B_ * T_ * D_ * 2);
  unsigned short* k_bf   = (unsigned short*)take((size_t)B_ * S_ * D_ * 2);
  unsigned short* vt_bf  = (unsigned short*)take((size_t)B_ * D_ * S_ * 2);
  unsigned short* qt_bf  = (unsigned short*)take((size_t)B_ * T_ * H_ * D_ * 2);
  unsigned short* ao_bf  = (unsigned short*)take((size_t)B_ * T_ * D_ * 2);
  unsigned short* wq_bf  = (unsigned short*)take((size_t)D_ * D_ * 2);
  unsigned short* wk_bf  = (unsigned short*)take((size_t)D_ * D_ * 2);
  unsigned short* wv_bf  = (unsigned short*)take((size_t)D_ * D_ * 2);
  unsigned short* wo_bf  = (unsigned short*)take((size_t)D_ * D_ * 2);
  unsigned short* wrt_bf = (unsigned short*)take((size_t)D_ * D_ * 2);
  float* Pbuf = (float*)take((size_t)T_ * B_ * H_ * S_ * 4);   // 64 MB

  // 1) precision conversion / packing
  {
    int n = B_ * S_ * D_;
    concat_kv_bf16<<<(n + 255) / 256, 256, 0, stream>>>(mem, x, kv_bf);
    int nw = D_ * D_;
    cvt_f32_bf16<<<(nw + 255) / 256, 256, 0, stream>>>(Wq, wq_bf, nw);
    cvt_f32_bf16<<<(nw + 255) / 256, 256, 0, stream>>>(Wk, wk_bf, nw);
    cvt_f32_bf16<<<(nw + 255) / 256, 256, 0, stream>>>(Wv, wv_bf, nw);
    cvt_f32_bf16<<<(nw + 255) / 256, 256, 0, stream>>>(Wo, wo_bf, nw);
    transpose_cvt_bf16<<<(nw + 255) / 256, 256, 0, stream>>>(Wr, wrt_bf);
  }

  // 2) projections (all NT, bf16 WMMA)
  // Q[b,t,:] = x[b,t,:] @ Wq^T  (x lives at kv rows M..S-1 per batch)
  gemm_nt_bf16<<<dim3(T_ / 16, D_ / 64, B_), 128, 0, stream>>>(
      kv_bf + (size_t)M_ * D_, D_, (long)S_ * D_,
      wq_bf, D_, 0,
      q_bf, D_, 1, (long)T_ * D_, D_, 0, nullptr);
  // K[b,s,:] = kv @ Wk^T
  gemm_nt_bf16<<<dim3(S_ / 16, D_ / 64, B_), 128, 0, stream>>>(
      kv_bf, D_, (long)S_ * D_,
      wk_bf, D_, 0,
      k_bf, D_, 1, (long)S_ * D_, D_, 0, nullptr);
  // Vt[b,d,s] = (kv @ Wv^T)[b,s,d]  (transposed store)
  gemm_nt_bf16<<<dim3(S_ / 16, D_ / 64, B_), 128, 0, stream>>>(
      kv_bf, D_, (long)S_ * D_,
      wv_bf, D_, 0,
      vt_bf, 1, S_, (long)D_ * S_, D_, 0, nullptr);
  // Qt[b,t,h,d] = sum_{d'} Q[b,t,h,d'] * Wr[h*64+d', d]  (per-head, K=64)
  gemm_nt_bf16<<<dim3((B_ * T_) / 16, D_ / 64, H_), 128, 0, stream>>>(
      q_bf, D_, HD_,
      wrt_bf, D_, HD_,
      qt_bf, (long)H_ * D_, 1, D_, HD_, 0, nullptr);

  // 3) relative-position scores: stream rel_pos once (HBM-bound step)
  pos_scores<<<T_, 256, 0, stream>>>(qt_bf, rel, Pbuf);

  // 4) fused attention (content WMMA + pos + online softmax + @V)
  attn_core<<<dim3(T_ / 16, H_, B_), 32, 0, stream>>>(q_bf, k_bf, vt_bf, Pbuf,
                                                      ao_bf);

  // 5) output projection with bias, f32 out
  gemm_nt_bf16<<<dim3((B_ * T_) / 16, D_ / 64, 1), 128, 0, stream>>>(
      ao_bf, D_, 0,
      wo_bf, D_, 0,
      out, D_, 1, 0, D_, 1, bo);
}
